// End_FC_Block_9852654977033
// MI455X (gfx1250) — matrix operand order repaired and checked by reading, not compiled, not run on hardware
//
#include <hip/hip_runtime.h>
#include <hip/hip_bf16.h>

// Problem constants (fixed by the reference)
#define B_ROWS 16384
#define D_IN   2048
#define D_OUT  1024
#define BN_EPS 1e-5f

typedef __attribute__((ext_vector_type(16))) _Float16 v16h;
typedef __attribute__((ext_vector_type(8)))  float    v8f;
typedef __attribute__((ext_vector_type(4)))  float    v4f;

// ---------------------------------------------------------------------------
// Kernel 1: binarize weights to f16 (+1 / -1). sign(w) is exact in f16.
// ---------------------------------------------------------------------------
__global__ __launch_bounds__(256)
void k_binarize(const float* __restrict__ w, _Float16* __restrict__ wb, int n) {
    int i = blockIdx.x * 256 + threadIdx.x;
    if (i < n) wb[i] = (w[i] >= 0.0f) ? (_Float16)1.0f : (_Float16)(-1.0f);
}

// ---------------------------------------------------------------------------
// Kernel 2: y = x @ sign(W)^T via V_WMMA_F32_16X16X32_F16.
//
// Workgroup = 256 threads = 8 waves in a 2(M) x 4(N) grid; WG tile 128x128.
// Per K-step (K=32):
//   - all 256 threads cooperatively load x[128][32] (f32), split each value
//     into f16 hi + f16 lo (lo = f32 residual -> near-fp32 GEMM accuracy,
//     weights +-1 are exact in f16), and store straight into LDS in WMMA
//     A-fragment layout (one conversion per element instead of one per
//     consuming wave);
//   - each wave ds_loads its A fragments (32B/lane contiguous) and keeps a
//     64(M) x 32(N) accumulator block (8 x v8f = 64 VGPRs, no VGPR-MSB
//     pressure), with B fragments (32B/lane contiguous) loaded directly from
//     the 4MB binarized-weight buffer (L2 resident).
//
// A-fragment (16x32 f16) layout used: lane -> M = lane&15; elements e ->
// K = 16*(lane>=16)+e (contiguous k order, same as B).
// B-fragment (32x16 f16): lane -> N = lane&15, elements -> K = 16*(lane>=16)+e;
// since B[k][n] = wb[n][k], that is 16 contiguous f16 in memory.
// C/D: VGPR r -> M = r + 8*(lane>=16), N = lane&15.
// ---------------------------------------------------------------------------
__global__ __launch_bounds__(256)
void k_gemm_wmma(const float* __restrict__ x, const _Float16* __restrict__ wb,
                 float* __restrict__ y) {
    __shared__ __align__(32) _Float16 Ahi[8][512];   // 8 subtiles * 32 lanes * 16
    __shared__ __align__(32) _Float16 Alo[8][512];   // 16KB total

    const int t    = threadIdx.x;          // 0..255
    const int lane = t & 31;               // lane within wave
    const int wid  = t >> 5;               // wave 0..7
    const int wm   = wid >> 2;             // 0..1  (M block of 64)
    const int wn   = wid & 3;              // 0..3  (N block of 32)
    const int half = lane >> 4;
    const int l15  = lane & 15;

    const int m0 = (blockIdx.x >> 3) * 128;   // 128 M-tiles
    const int n0 = (blockIdx.x & 7) * 128;    // 8   N-tiles

    // Staging role: thread t handles A subtile `ssub` = t>>5, fragment lane
    // `slane` = t&31 (same mapping as the consuming wave's lane).
    const int ssub   = t >> 5;
    const int slane  = t & 31;
    const int s_half = slane >> 4;
    const int s_l15  = slane & 15;
    const float* arow = x + (size_t)(m0 + ssub * 16 + s_l15) * D_IN + s_half * 16;

    const int b_koff = half * 16;
    const _Float16* bp0 = wb + (size_t)(n0 + wn * 32 + l15) * D_IN + b_koff;
    const _Float16* bp1 = bp0 + (size_t)16 * D_IN;

    v8f acc[4][2] = {};                    // [a-subtile][b-subtile]

    for (int k0 = 0; k0 < D_IN; k0 += 32) {
        // ---- Stage A hi/lo fragments into LDS (one conversion per element).
        {
            const float* ap = arow + k0;
            __builtin_prefetch(ap + 32, 0, 0);     // next K-block of this row
            v4f xv4[4];
            xv4[0] = *(const v4f*)(ap + 0);
            xv4[1] = *(const v4f*)(ap + 4);
            xv4[2] = *(const v4f*)(ap + 8);
            xv4[3] = *(const v4f*)(ap + 12);
            v16h hi, lo;
#pragma unroll
            for (int e = 0; e < 16; ++e) {
                float f = xv4[e >> 2][e & 3];
                _Float16 h = (_Float16)f;
                hi[e] = h;
                lo[e] = (_Float16)(f - (float)h);  // f32 residual
            }
            *(v16h*)&Ahi[ssub][slane * 16] = hi;   // 32B per thread, contiguous
            *(v16h*)&Alo[ssub][slane * 16] = lo;
        }
        __syncthreads();

        // ---- B fragments: 2 n-subtiles, 32B contiguous per lane from global.
        v16h Bf0 = *(const v16h*)(bp0 + k0);
        v16h Bf1 = *(const v16h*)(bp1 + k0);

        // ---- 4 a-subtiles x 2 b-subtiles x (hi+lo) = 16 WMMAs per wave.
#pragma unroll
        for (int a = 0; a < 4; ++a) {
            const int sub = wm * 4 + a;
            v16h ahi = *(const v16h*)&Ahi[sub][lane * 16];
            v16h alo = *(const v16h*)&Alo[sub][lane * 16];
            acc[a][0] = __builtin_amdgcn_wmma_f32_16x16x32_f16(
                false, ahi, false, Bf0, (short)0, acc[a][0], false, false);
            acc[a][0] = __builtin_amdgcn_wmma_f32_16x16x32_f16(
                false, alo, false, Bf0, (short)0, acc[a][0], false, false);
            acc[a][1] = __builtin_amdgcn_wmma_f32_16x16x32_f16(
                false, ahi, false, Bf1, (short)0, acc[a][1], false, false);
            acc[a][1] = __builtin_amdgcn_wmma_f32_16x16x32_f16(
                false, alo, false, Bf1, (short)0, acc[a][1], false, false);
        }
        __syncthreads();   // LDS reuse next iteration
    }

    // ---- Store C/D: VGPR r -> M = r + 8*half, N = l15 (coalesced per half).
#pragma unroll
    for (int a = 0; a < 4; ++a)
#pragma unroll
        for (int b = 0; b < 2; ++b)
#pragma unroll
            for (int r = 0; r < 8; ++r) {
                int m = m0 + wm * 64 + a * 16 + half * 8 + r;
                int n = n0 + wn * 32 + b * 16 + l15;
                y[(size_t)m * D_OUT + n] = acc[a][b][r];
            }
}

// ---------------------------------------------------------------------------
// Kernel 3: per-column partial sums / sums of squares over a 128-row block.
// Fully coalesced; deterministic (no atomics).
// ---------------------------------------------------------------------------
__global__ __launch_bounds__(256)
void k_col_partial(const float* __restrict__ y, float* __restrict__ psum,
                   float* __restrict__ psq) {
    const int bb = blockIdx.x;             // 128 row-blocks of 128 rows
    const int t  = threadIdx.x;            // 256 threads; cols t, t+256, ...
    float s0 = 0.f, s1 = 0.f, s2 = 0.f, s3 = 0.f;
    float q0 = 0.f, q1 = 0.f, q2 = 0.f, q3 = 0.f;
    const float* base = y + (size_t)bb * 128 * D_OUT;
    for (int r = 0; r < 128; ++r) {
        const float* row = base + r * D_OUT;
        float v0 = row[t + 0];   s0 += v0; q0 += v0 * v0;
        float v1 = row[t + 256]; s1 += v1; q1 += v1 * v1;
        float v2 = row[t + 512]; s2 += v2; q2 += v2 * v2;
        float v3 = row[t + 768]; s3 += v3; q3 += v3 * v3;
    }
    float* ps = psum + (size_t)bb * D_OUT;
    float* pq = psq  + (size_t)bb * D_OUT;
    ps[t + 0] = s0; ps[t + 256] = s1; ps[t + 512] = s2; ps[t + 768] = s3;
    pq[t + 0] = q0; pq[t + 256] = q1; pq[t + 512] = q2; pq[t + 768] = q3;
}

// ---------------------------------------------------------------------------
// Kernel 4: reduce partials -> per-column BN scale/shift.
// out = y*scale + shift, scale = gamma*rsqrt(var+eps), shift = beta - mean*scale
// (the linear bias cancels under BatchNorm and is skipped entirely).
// ---------------------------------------------------------------------------
__global__ __launch_bounds__(256)
void k_bn_stats(const float* __restrict__ psum, const float* __restrict__ psq,
                const float* __restrict__ gamma, const float* __restrict__ beta,
                float* __restrict__ scale, float* __restrict__ shiftv) {
    int o = blockIdx.x * 256 + threadIdx.x;       // 0..1023
    float s = 0.f, q = 0.f;
    for (int p = 0; p < 128; ++p) {
        s += psum[(size_t)p * D_OUT + o];
        q += psq [(size_t)p * D_OUT + o];
    }
    const float inv_n = 1.0f / (float)B_ROWS;
    float mean = s * inv_n;
    float var  = q * inv_n - mean * mean;         // biased variance
    float sc   = gamma[o] * rsqrtf(var + BN_EPS);
    scale[o]  = sc;
    shiftv[o] = beta[o] - mean * sc;
}

// ---------------------------------------------------------------------------
// Kernel 5: in-place fused BN apply, float4 vectorized.
// ---------------------------------------------------------------------------
__global__ __launch_bounds__(256)
void k_bn_apply(float* __restrict__ y, const float* __restrict__ scale,
                const float* __restrict__ shiftv) {
    size_t i = (size_t)blockIdx.x * 256 + threadIdx.x;  // float4 index
    int cg = (int)(i & (D_OUT / 4 - 1));                // column group (0..255)
    float4 v  = ((float4*)y)[i];
    float4 sc = ((const float4*)scale)[cg];
    float4 sh = ((const float4*)shiftv)[cg];
    v.x = v.x * sc.x + sh.x;
    v.y = v.y * sc.y + sh.y;
    v.z = v.z * sc.z + sh.z;
    v.w = v.w * sc.w + sh.w;
    ((float4*)y)[i] = v;
}

// ---------------------------------------------------------------------------
extern "C" void kernel_launch(void* const* d_in, const int* in_sizes, int n_in,
                              void* d_out, int out_size, void* d_ws, size_t ws_size,
                              hipStream_t stream) {
    const float* x     = (const float*)d_in[0];
    const float* w     = (const float*)d_in[1];
    // d_in[2] = bias: mathematically cancelled by BatchNorm (per-column
    // constant shift); unused.
    const float* gamma = (const float*)d_in[3];
    const float* beta  = (const float*)d_in[4];
    float* y = (float*)d_out;                     // y lives in d_out, BN in place

    // Workspace layout (< 6 MB):
    //   [0, 4MB)            : binarized weights, f16 [D_OUT][D_IN]
    //   [4MB, 4.5MB)        : psum  [128][D_OUT]
    //   [4.5MB, 5MB)        : psq   [128][D_OUT]
    //   [5MB, +4KB)         : scale [D_OUT]
    //   [5MB+4KB, +8KB)     : shift [D_OUT]
    char* ws = (char*)d_ws;
    _Float16* wb   = (_Float16*)ws;
    float* psum    = (float*)(ws + (size_t)4 * 1024 * 1024);
    float* psq     = (float*)(ws + (size_t)4 * 1024 * 1024 + 512 * 1024);
    float* scale   = (float*)(ws + (size_t)5 * 1024 * 1024);
    float* shiftv  = (float*)(ws + (size_t)5 * 1024 * 1024 + 4096);

    const int wn = D_OUT * D_IN;                                   // 2M
    k_binarize<<<(wn + 255) / 256, 256, 0, stream>>>(w, wb, wn);

    const int gtiles = (B_ROWS / 128) * (D_OUT / 128);             // 1024
    k_gemm_wmma<<<gtiles, 256, 0, stream>>>(x, wb, y);

    k_col_partial<<<B_ROWS / 128, 256, 0, stream>>>(y, psum, psq); // 128 blocks

    k_bn_stats<<<D_OUT / 256, 256, 0, stream>>>(psum, psq, gamma, beta,
                                                scale, shiftv);

    const size_t nvec4 = (size_t)B_ROWS * D_OUT / 4;               // 4M
    k_bn_apply<<<(unsigned)(nvec4 / 256), 256, 0, stream>>>(y, scale, shiftv);
}
